// NLayerMambaModel_41223096107582
// MI455X (gfx1250) — matrix operand order repaired and checked by reading, not compiled, not run on hardware
//
#include <hip/hip_runtime.h>
#include <hip/hip_bf16.h>
#include <math.h>
#include <stdint.h>

// ---------------- problem constants (match reference) ----------------
#define BB      2
#define LL      1024
#define DMODEL  1024
#define DIN     2048          // d_inner
#define NST     16            // d_state
#define DTR     64            // dt_rank
#define DCONV   4
#define MM      (BB * LL)     // 2048 flattened rows

typedef __attribute__((ext_vector_type(16))) _Float16 v16h;
typedef __attribute__((ext_vector_type(8)))  _Float16 v8h;
typedef __attribute__((ext_vector_type(8)))  float    v8f;

union HalfPack { v16h v; v8h h[2]; };

// =====================================================================
// Generic f16 WMMA GEMM:  C[M,N] = A[M,K] * W[N,K]^T   (f32 accumulate)
//   mode 0: store raw
//   mode 1: c = softplus(c + extra[n])           (dt projection)
//   mode 2: c = c + extra[m*N + n]               (residual add)
// Workgroup: 256 threads = 8 waves; tile 128(M) x 64(N); K step 32.
// B tile staged to LDS via CDNA5 async copy (ASYNCcnt), double buffered:
// the DMA for tile i+1 overlaps the 4 WMMAs of tile i.
// Requires: M % 128 == 0, K % 32 == 0. N may be ragged (guarded).
// =====================================================================
__global__ __launch_bounds__(256)
void gemm_f16_wmma(const _Float16* __restrict__ A,
                   const _Float16* __restrict__ Bw,
                   float* __restrict__ C,
                   int M, int N, int K,
                   int mode, const float* __restrict__ extra)
{
    __shared__ __attribute__((aligned(64))) _Float16 Bs[2][64][32]; // 2 x 4KB

    const int tid   = threadIdx.x;
    const int wave  = tid >> 5;       // 0..7
    const int lane  = tid & 31;
    const int msub  = lane & 15;
    const int khalf = lane >> 4;      // 0/1

    const int block_m = blockIdx.y * 128;
    const int block_n = blockIdx.x * 64;
    const int arow    = block_m + wave * 16 + msub;   // this lane's A row

    // per-thread B staging coordinates: one 16B chunk each
    const int  n_local = tid >> 2;          // 0..63
    const int  k_chunk = (tid & 3) * 8;     // 0,8,16,24
    const int  gn_st   = block_n + n_local;
    const bool in_n    = (gn_st < N);

    const int ntiles = K / 32;

    // issue async global->LDS copy of B tile `tile` into buffer `buf`
    auto stage = [&](int tile, int buf) {
        if (in_n) {
            unsigned lds_off = (unsigned)(uintptr_t)&Bs[buf][n_local][k_chunk];
            unsigned goff    = (unsigned)(((size_t)gn_st * K + tile * 32 + k_chunk)
                                          * sizeof(_Float16));
            asm volatile("global_load_async_to_lds_b128 %0, %1, %2"
                         :: "v"(lds_off), "v"(goff), "s"(Bw) : "memory");
        } else {
            v8h z = {};
            *(v8h*)&Bs[buf][n_local][k_chunk] = z;     // zero-pad ragged N
        }
    };

    stage(0, 0);

    v8f acc[4] = {};                  // four 16x16 N-subtiles

    for (int i = 0; i < ntiles; ++i) {
        const int cur = i & 1;

        // kick off next tile's DMA into the other buffer (overlaps compute)
        if (i + 1 < ntiles) {
            stage(i + 1, cur ^ 1);
            asm volatile("s_wait_asynccnt 0x1" ::: "memory"); // tile i done
        } else {
            asm volatile("s_wait_asynccnt 0x0" ::: "memory");
        }
        __syncthreads();              // tile i visible to all waves

        // A fragment, contiguous k order (matches B):
        // lane (khalf=0): elems 0..15 = K[0..15]
        // lane (khalf=1): elems 0..15 = K[16..31]
        const _Float16* ap = A + (size_t)arow * K + i * 32;
        HalfPack af;
        af.h[0] = *(const v8h*)(ap + 16 * khalf);
        af.h[1] = *(const v8h*)(ap + 16 * khalf + 8);
        const v16h afrag = af.v;

        #pragma unroll
        for (int t = 0; t < 4; ++t) {
            // B 32x16 layout: lane n (group khalf) holds K[16*khalf .. +15]
            const v16h bfrag = *(const v16h*)(&Bs[cur][t * 16 + msub][khalf * 16]);
            acc[t] = __builtin_amdgcn_wmma_f32_16x16x32_f16(
                         false, afrag, false, bfrag,
                         (short)0, acc[t], false, false);
        }
        __syncthreads();  // everyone done reading Bs[cur] before it is re-staged
    }

    // ---- epilogue + store (C/D layout: VGPR r -> row 8*khalf + r)
    const int gm0 = block_m + wave * 16 + 8 * khalf;
    const int gn0 = block_n + msub;
    #pragma unroll
    for (int t = 0; t < 4; ++t) {
        const int gn = gn0 + t * 16;
        if (gn >= N) continue;
        #pragma unroll
        for (int r = 0; r < 8; ++r) {
            const int gm = gm0 + r;
            float c = acc[t][r];
            if (mode == 1) {                       // bias + softplus
                const float v = c + extra[gn];
                c = (v > 20.f) ? v : log1pf(__expf(v));
            } else if (mode == 2) {                // residual add
                c += extra[(size_t)gm * N + gn];
            }
            C[(size_t)gm * N + gn] = c;
        }
    }
}

// =====================================================================
// f32 -> f16 cast (contiguous)
// =====================================================================
__global__ void cast_f32_to_f16(const float* __restrict__ in,
                                _Float16* __restrict__ out, int n)
{
    int i = blockIdx.x * blockDim.x + threadIdx.x;
    if (i < n) out[i] = (_Float16)in[i];
}

// f32 -> f16 cast of x_dbl[:, 0:64] (row stride 96)
__global__ void cast_dtlr(const float* __restrict__ xdbl,
                          _Float16* __restrict__ out)
{
    int i = blockIdx.x * blockDim.x + threadIdx.x;   // over MM*64
    if (i >= MM * DTR) return;
    int m = i / DTR, r = i % DTR;
    out[i] = (_Float16)xdbl[(size_t)m * (DTR + 2 * NST) + r];
}

// =====================================================================
// Causal depthwise conv (width 4) + bias + SiLU; reads u_raw = xz[:, :DIN]
// =====================================================================
__global__ void conv_silu_kernel(const float* __restrict__ xz,
                                 const float* __restrict__ cw,   // [DIN,4]
                                 const float* __restrict__ cb,   // [DIN]
                                 float* __restrict__ u,
                                 _Float16* __restrict__ u16)
{
    int idx = blockIdx.x * blockDim.x + threadIdx.x; // over MM*DIN
    if (idx >= MM * DIN) return;
    const int d  = idx % DIN;
    const int ml = idx / DIN;       // b*L + l
    const int l  = ml % LL;

    float acc = cb[d];
    #pragma unroll
    for (int j = 0; j < DCONV; ++j) {
        const int ls = l - (DCONV - 1) + j;
        if (ls >= 0)
            acc += cw[d * DCONV + j] *
                   xz[(size_t)(ml - (DCONV - 1) + j) * (2 * DIN) + d];
    }
    const float s = acc / (1.f + __expf(-acc));      // silu
    u[idx]   = s;
    u16[idx] = (_Float16)s;
}

// =====================================================================
// Selective scan: one thread per (b, d) channel, serial over L.
// h[n] = exp(dt*A[n])*h[n] + dt*B[n]*u ;  y = <h,C> + D*u ; y *= silu(z)
// =====================================================================
__global__ void scan_kernel(const float* __restrict__ u,
                            const float* __restrict__ dt,
                            const float* __restrict__ xdbl,   // [MM, 96]
                            const float* __restrict__ A_log,  // [DIN, 16]
                            const float* __restrict__ Dvec,   // [DIN]
                            const float* __restrict__ xz,     // z at col 2048+d
                            float* __restrict__ y,
                            _Float16* __restrict__ y16)
{
    const int ch = blockIdx.x * blockDim.x + threadIdx.x;  // 0..BB*DIN-1
    if (ch >= BB * DIN) return;
    const int b = ch / DIN;
    const int d = ch % DIN;

    float Aneg[NST];
    #pragma unroll
    for (int n = 0; n < NST; ++n) Aneg[n] = -__expf(A_log[d * NST + n]);
    const float Dv = Dvec[d];

    float h[NST];
    #pragma unroll
    for (int n = 0; n < NST; ++n) h[n] = 0.f;

    for (int l = 0; l < LL; ++l) {
        const size_t m  = (size_t)b * LL + l;
        const float dtv = dt[m * DIN + d];
        const float uv  = u[m * DIN + d];

        // B/C blocks: 32 contiguous floats at row offset DTR (16B aligned)
        float bc[2 * NST];
        const float4* p = (const float4*)(xdbl + m * (DTR + 2 * NST) + DTR);
        #pragma unroll
        for (int q = 0; q < 8; ++q) ((float4*)bc)[q] = p[q];

        float yv = 0.f;
        #pragma unroll
        for (int n = 0; n < NST; ++n) {
            const float dA = __expf(dtv * Aneg[n]);
            h[n] = dA * h[n] + dtv * bc[n] * uv;           // B block
            yv  += h[n] * bc[NST + n];                     // C block
        }
        yv += Dv * uv;

        const float z = xz[m * (2 * DIN) + DIN + d];
        yv *= z / (1.f + __expf(-z));                      // * silu(z)

        y[m * DIN + d]   = yv;
        y16[m * DIN + d] = (_Float16)yv;
    }
}

// =====================================================================
// host side
// =====================================================================
static inline size_t alignup(size_t x) { return (x + 255) & ~(size_t)255; }

extern "C" void kernel_launch(void* const* d_in, const int* in_sizes, int n_in,
                              void* d_out, int out_size, void* d_ws, size_t ws_size,
                              hipStream_t stream)
{
    (void)in_sizes; (void)n_in; (void)out_size; (void)ws_size;

    const float* x      = (const float*)d_in[0];   // [2,1024,1024]
    const float* W_in   = (const float*)d_in[1];   // [2,4096,1024]
    const float* conv_w = (const float*)d_in[2];   // [2,2048,4]
    const float* conv_b = (const float*)d_in[3];   // [2,2048]
    const float* W_x    = (const float*)d_in[4];   // [2,96,2048]
    const float* W_dt   = (const float*)d_in[5];   // [2,2048,64]
    const float* b_dt   = (const float*)d_in[6];   // [2,2048]
    const float* A_log  = (const float*)d_in[7];   // [2,2048,16]
    const float* Dp     = (const float*)d_in[8];   // [2,2048]
    const float* W_out  = (const float*)d_in[9];   // [2,1024,2048]
    float* outp = (float*)d_out;

    // ---- workspace carve-up
    char* ws = (char*)d_ws; size_t off = 0;
    auto carve = [&](size_t bytes) { void* p = ws + off; off = alignup(off + bytes); return p; };

    _Float16* h16    = (_Float16*)carve((size_t)MM * DMODEL * 2);
    _Float16* Win16  = (_Float16*)carve((size_t)2 * DIN * DMODEL * 2);
    _Float16* Wx16   = (_Float16*)carve((size_t)(DTR + 2 * NST) * DIN * 2);
    _Float16* Wdt16  = (_Float16*)carve((size_t)DIN * DTR * 2);
    _Float16* Wout16 = (_Float16*)carve((size_t)DMODEL * DIN * 2);
    float*    xzb    = (float*)carve((size_t)MM * 2 * DIN * 4);
    float*    ub     = (float*)carve((size_t)MM * DIN * 4);
    _Float16* u16    = (_Float16*)carve((size_t)MM * DIN * 2);
    float*    xdbl   = (float*)carve((size_t)MM * (DTR + 2 * NST) * 4);
    _Float16* dtlr16 = (_Float16*)carve((size_t)MM * DTR * 2);
    float*    dtb    = (float*)carve((size_t)MM * DIN * 4);
    float*    yb     = (float*)carve((size_t)MM * DIN * 4);
    _Float16* y16    = (_Float16*)carve((size_t)MM * DIN * 2);
    float*    h1     = (float*)carve((size_t)MM * DMODEL * 4);

    const int CT = 256;
    auto cgrid = [](int n) { return (n + 255) / 256; };

    // layer input (f16 GEMM operand)
    cast_f32_to_f16<<<cgrid(MM * DMODEL), CT, 0, stream>>>(x, h16, MM * DMODEL);

    for (int l = 0; l < 2; ++l) {
        const size_t oWin  = (size_t)l * 2 * DIN * DMODEL;
        const size_t oWx   = (size_t)l * (DTR + 2 * NST) * DIN;
        const size_t oWdt  = (size_t)l * DIN * DTR;
        const size_t oWout = (size_t)l * DMODEL * DIN;

        cast_f32_to_f16<<<cgrid(2 * DIN * DMODEL), CT, 0, stream>>>(W_in + oWin,  Win16,  2 * DIN * DMODEL);
        cast_f32_to_f16<<<cgrid((DTR + 2 * NST) * DIN), CT, 0, stream>>>(W_x + oWx, Wx16, (DTR + 2 * NST) * DIN);
        cast_f32_to_f16<<<cgrid(DIN * DTR), CT, 0, stream>>>(W_dt + oWdt, Wdt16, DIN * DTR);
        cast_f32_to_f16<<<cgrid(DMODEL * DIN), CT, 0, stream>>>(W_out + oWout, Wout16, DMODEL * DIN);

        // xz = h * W_in^T : [2048, 4096]
        gemm_f16_wmma<<<dim3(2 * DIN / 64, MM / 128), 256, 0, stream>>>(
            h16, Win16, xzb, MM, 2 * DIN, DMODEL, 0, nullptr);

        // depthwise causal conv + silu -> u
        conv_silu_kernel<<<cgrid(MM * DIN), CT, 0, stream>>>(
            xzb, conv_w + (size_t)l * DIN * DCONV, conv_b + (size_t)l * DIN, ub, u16);

        // x_dbl = u * W_x^T : [2048, 96]
        gemm_f16_wmma<<<dim3((DTR + 2 * NST + 63) / 64, MM / 128), 256, 0, stream>>>(
            u16, Wx16, xdbl, MM, DTR + 2 * NST, DIN, 0, nullptr);

        // dt = softplus(dt_lowrank * W_dt^T + b_dt) : [2048, 2048]
        cast_dtlr<<<cgrid(MM * DTR), CT, 0, stream>>>(xdbl, dtlr16);
        gemm_f16_wmma<<<dim3(DIN / 64, MM / 128), 256, 0, stream>>>(
            dtlr16, Wdt16, dtb, MM, DIN, DTR, 1, b_dt + (size_t)l * DIN);

        // selective scan + D skip + silu(z) gating
        scan_kernel<<<(BB * DIN) / 256, 256, 0, stream>>>(
            ub, dtb, xdbl, A_log + (size_t)l * DIN * NST, Dp + (size_t)l * DIN,
            xzb, yb, y16);

        // out-proj; final layer adds the global residual (x) and writes d_out
        float* Cdst = (l == 1) ? outp : h1;
        gemm_f16_wmma<<<dim3(DMODEL / 64, MM / 128), 256, 0, stream>>>(
            y16, Wout16, Cdst, MM, DMODEL, DIN, (l == 1) ? 2 : 0,
            (l == 1) ? x : nullptr);

        if (l == 0)
            cast_f32_to_f16<<<cgrid(MM * DMODEL), CT, 0, stream>>>(h1, h16, MM * DMODEL);
    }
}
